// UNetBlockDownsasa_unique_69501160783981
// MI455X (gfx1250) — matrix operand order repaired and checked by reading, not compiled, not run on hardware
//
#include <hip/hip_runtime.h>

typedef __attribute__((ext_vector_type(16))) _Float16 v16h;
typedef __attribute__((ext_vector_type(8)))  float    v8f;

#define HWPIX (128*128)
#define CCH   128
#define NB    4
#define NHEADS 4
#define DHEAD 32
#define QSCALE 0.17677669529663687f   // 32^-0.5

// ---------------------------------------------------------------------------
// Projection GEMM: out[b, head, p, d] = (sum_c W[o][c]*x[b,c,p] + bias[o]) * scale
// One wave = 16-pixel x 128-outchannel tile via v_wmma_f32_16x16x32_f16.
// ---------------------------------------------------------------------------
__global__ __launch_bounds__(128)
void proj_gemm(const float* __restrict__ x, const float* __restrict__ Wm,
               const float* __restrict__ bias, float* __restrict__ out,
               float scale)
{
    // W[o][c] staged as f16, row pitch 130 halves -> conflict-free LDS reads
    __shared__ _Float16 wl[128 * 130];
    const int tid = threadIdx.x;
    for (int i = tid; i < 128 * 128; i += 128) {
        int o = i >> 7, c = i & 127;
        wl[o * 130 + c] = (_Float16)Wm[i];
    }
    __syncthreads();

    const int wave = tid >> 5;
    const int lane = tid & 31;
    const int tile = blockIdx.x * 4 + wave;          // 0..4095
    const int p0   = tile << 4;                      // flat pixel over B*HW
    const int b    = p0 / HWPIX;
    const int pix0 = p0 - b * HWPIX;

    const int m      = lane & 15;                    // M row (pixel within tile)
    const int khalf  = (lane >> 4) << 3;             // A: K sub-block select (0/8)
    const int koff   = (lane >> 4) << 3;             // B: K sub-block select (0/8)
    const float* xb  = x + (size_t)b * CCH * HWPIX + (pix0 + m);

    v8f acc[8];
    #pragma unroll
    for (int nt = 0; nt < 8; ++nt)
        #pragma unroll
        for (int j = 0; j < 8; ++j) acc[nt][j] = 0.0f;

    #pragma unroll
    for (int k0 = 0; k0 < CCH; k0 += 32) {
        // --- A fragment: 16x32 f16, ISA lane layout ---
        v16h a;
        #pragma unroll
        for (int j = 0; j < 4; ++j) {
            int c = k0 + khalf + 2 * j;
            a[2 * j]     = (_Float16)xb[(size_t)c       * HWPIX];
            a[2 * j + 1] = (_Float16)xb[(size_t)(c + 1) * HWPIX];
        }
        #pragma unroll
        for (int j = 4; j < 8; ++j) {
            int c = k0 + 16 + khalf + 2 * (j - 4);
            a[2 * j]     = (_Float16)xb[(size_t)c       * HWPIX];
            a[2 * j + 1] = (_Float16)xb[(size_t)(c + 1) * HWPIX];
        }
        // --- 8 N-tiles, reuse A ---
        #pragma unroll
        for (int nt = 0; nt < 8; ++nt) {
            int o = nt * 16 + (lane & 15);           // output channel (N col)
            const _Float16* wp = &wl[o * 130 + k0 + koff];
            v16h bf;
            #pragma unroll
            for (int j = 0; j < 16; ++j) bf[j] = wp[j < 8 ? j : j + 8];
            acc[nt] = __builtin_amdgcn_wmma_f32_16x16x32_f16(
                false, a, false, bf, (short)0, acc[nt], false, false);
        }
    }

    // epilogue: C/D layout -> N = lane&15, M = v + 8*(lane>>4)
    const int mbase = (lane >> 4) << 3;
    #pragma unroll
    for (int nt = 0; nt < 8; ++nt) {
        int oc   = nt * 16 + (lane & 15);
        int head = oc >> 5, d = oc & 31;
        float bia = bias[oc];
        #pragma unroll
        for (int v = 0; v < 8; ++v) {
            int p = pix0 + mbase + v;
            out[(((size_t)b * NHEADS + head) * HWPIX + p) * DHEAD + d] =
                (acc[nt][v] + bia) * scale;
        }
    }
}

// ---------------------------------------------------------------------------
// SASA attention epilogue: one thread per (b, head, pixel).
// Border taps == bias vector (pad -> conv1x1 -> unfold identity).
// ---------------------------------------------------------------------------
__global__ __launch_bounds__(256)
void sasa_attn(const float* __restrict__ Q, const float* __restrict__ KV1,
               const float* __restrict__ KV2, const float* __restrict__ bkv,
               float* __restrict__ out)
{
    const int gid  = blockIdx.x * 256 + threadIdx.x;   // (b*4+head)*HW + p
    const int p    = gid & (HWPIX - 1);
    const int bh   = gid >> 14;                        // HW = 2^14
    const int head = bh & 3;
    const int b    = bh >> 2;
    const int y    = p >> 7, xq = p & 127;

    float ql[32], bv[32], ov[32];
    const float* qp = Q + (size_t)gid * DHEAD;
    #pragma unroll
    for (int d = 0; d < 32; ++d) ql[d] = qp[d];
    #pragma unroll
    for (int d = 0; d < 32; ++d) bv[d] = bkv[head * 32 + d];
    #pragma unroll
    for (int d = 0; d < 32; ++d) ov[d] = 0.0f;

    float qdotb = 0.0f;
    #pragma unroll
    for (int d = 0; d < 32; ++d) qdotb += ql[d] * bv[d];

    #pragma unroll
    for (int br = 0; br < 2; ++br) {
        const float* kb = (br == 0 ? KV1 : KV2) + (size_t)bh * HWPIX * DHEAD;
        float lg[9];
        #pragma unroll
        for (int t = 0; t < 9; ++t) {
            int yy = y + t / 3 - 1, xx = xq + t % 3 - 1;
            if (yy >= 0 && yy < 128 && xx >= 0 && xx < 128) {
                const float* kp = kb + ((size_t)(yy * 128 + xx)) * DHEAD;
                float s = 0.0f;
                #pragma unroll
                for (int d = 0; d < 32; ++d) s += ql[d] * kp[d];
                lg[t] = s;
            } else {
                lg[t] = qdotb;
            }
        }
        float mx = lg[0];
        #pragma unroll
        for (int t = 1; t < 9; ++t) mx = fmaxf(mx, lg[t]);
        float ss = 0.0f;
        #pragma unroll
        for (int t = 0; t < 9; ++t) { lg[t] = __expf(lg[t] - mx); ss += lg[t]; }
        float inv = 1.0f / ss;
        #pragma unroll
        for (int t = 0; t < 9; ++t) {
            float w = lg[t] * inv;
            int yy = y + t / 3 - 1, xx = xq + t % 3 - 1;
            if (yy >= 0 && yy < 128 && xx >= 0 && xx < 128) {
                const float* kp = kb + ((size_t)(yy * 128 + xx)) * DHEAD;
                #pragma unroll
                for (int d = 0; d < 32; ++d) ov[d] += w * kp[d];
            } else {
                #pragma unroll
                for (int d = 0; d < 32; ++d) ov[d] += w * bv[d];
            }
        }
    }

    // out[b, head*32+d, y, x]
    float* op = out + ((size_t)b * 128 + head * 32) * HWPIX + p;
    #pragma unroll
    for (int d = 0; d < 32; ++d) op[(size_t)d * HWPIX] = 0.5f * ov[d];
}

// ---------------------------------------------------------------------------
extern "C" void kernel_launch(void* const* d_in, const int* in_sizes, int n_in,
                              void* d_out, int out_size, void* d_ws, size_t ws_size,
                              hipStream_t stream) {
    const float* kvmap1 = (const float*)d_in[0];
    const float* qmap   = (const float*)d_in[1];
    const float* kvmap2 = (const float*)d_in[2];
    const float* Wq     = (const float*)d_in[3];
    const float* bq     = (const float*)d_in[4];
    const float* Wkv    = (const float*)d_in[5];
    const float* bkv    = (const float*)d_in[6];
    float* out = (float*)d_out;
    float* ws  = (float*)d_ws;

    const size_t per = (size_t)NB * NHEADS * HWPIX * DHEAD;  // 8,388,608 floats (32 MB)
    float* Qp = ws;
    float* K1 = ws + per;
    float* K2 = ws + 2 * per;

    const int tiles  = (NB * HWPIX) / 16;   // 4096 M-tiles
    const int blocks = tiles / 4;           // 4 waves per block
    proj_gemm<<<blocks, 128, 0, stream>>>(qmap,   Wq,  bq,  Qp, QSCALE);
    proj_gemm<<<blocks, 128, 0, stream>>>(kvmap1, Wkv, bkv, K1, 1.0f);
    proj_gemm<<<blocks, 128, 0, stream>>>(kvmap2, Wkv, bkv, K2, 1.0f);

    const int nthr = NB * NHEADS * HWPIX;   // 262144
    sasa_attn<<<nthr / 256, 256, 0, stream>>>(Qp, K1, K2, bkv, out);
}
